// MultiHeadedAttentionBias_25640954757159
// MI455X (gfx1250) — matrix operand order repaired and checked by reading, not compiled, not run on hardware
//
#include <hip/hip_runtime.h>
#include <hip/hip_bf16.h>

// MultiHeadedAttentionBias for MI455X (gfx1250, wave32, WMMA).
// Shapes: B=4, S=1024, D=512, H=8, DK=64.
// d_in order: query(0,unused), key(1,unused), value(2), bias(3), mask(4),
//             W_v(5), b_v(6), W_o(7), b_o(8)
// d_out = [ out (B*S*D f32) | bias copy (B*H*S*S f32) ]
//
// Roofline: ~8.6 GFLOP vs ~420 MB of HBM traffic dominated by the single-use
// 134 MB bias read + 134 MB bias copy-out. Strategy: touch bias exactly once
// (fused copy+mask+softmax+PV), keep p in LDS only, mark the bias stream
// non-temporal so L2 (192 MB) retains the re-used vbhT/mask/weights.

typedef __attribute__((ext_vector_type(16))) __bf16 v16bf;
typedef __attribute__((ext_vector_type(8)))  __bf16 v8bf;
typedef __attribute__((ext_vector_type(8)))  float  v8f;
typedef __attribute__((ext_vector_type(4)))  float  v4f;

#define HH 8
#define SS 1024
#define DD 512
#define DK 64
#define BB 4
#define NEGBIG (-1e9f)

// ---- WMMA fragment loaders (16-bit operands, contiguous k order, same in A and B) ----
// A (16x32 bf16): lane M = lane&15; lanes<16 hold K=0..15,
// lanes>=16 hold K=16..31 (same k order as B).
__device__ __forceinline__ v16bf load_a_frag(const __bf16* __restrict__ A,
                                             int ldk, int m, int kk, int lane) {
  int kbase = kk + ((lane & 16) ? 16 : 0);
  const __bf16* p = A + (size_t)m * ldk + kbase;
  v16bf a = *(const v16bf*)(p);
  return a;
}

// B (32x16 bf16): lane N = lane&15; lanes<16 hold K=0..15, lanes>=16 K=16..31.
// With B stored K-contiguous per column, this is ONE 32-byte load.
__device__ __forceinline__ v16bf load_b_frag(const __bf16* __restrict__ Bcol,
                                             int kk, int lane) {
  int khalf = kk + ((lane & 16) ? 16 : 0);
  return *(const v16bf*)(Bcol + khalf);
}

// ---------------- f32 -> bf16 convert ----------------
__global__ void f32_to_bf16(const float* __restrict__ src, __bf16* __restrict__ dst, int n) {
  int i = blockIdx.x * blockDim.x + threadIdx.x;
  int stride = gridDim.x * blockDim.x;
  for (; i < n; i += stride) dst[i] = (__bf16)src[i];
}

// ---------------- V projection: vproj = value @ W_v^T + b_v ----------------
// A = value_bf [4096 x 512], Bmat[k][n] = W_v[n][k] (W_v rows are K-contiguous).
// Epilogue writes transposed bf16 V: vbhT[((b*H+h)*DK + d)*S + s].
__global__ void __launch_bounds__(128)
vproj_gemm(const __bf16* __restrict__ Abf, const __bf16* __restrict__ Wvbf,
           const float* __restrict__ b_v, __bf16* __restrict__ vbhT) {
  int lane = threadIdx.x & 31;
  int wave = threadIdx.x >> 5;           // 4 waves -> 4 N-tiles of 16
  int m0 = blockIdx.x * 16;
  int ncol = blockIdx.y * 64 + wave * 16 + (lane & 15);
  int m = m0 + (lane & 15);
  const __bf16* Bcol = Wvbf + (size_t)ncol * DD;
  v8f acc = {};
#pragma unroll 4
  for (int kk = 0; kk < DD; kk += 32) {
    v16bf a = load_a_frag(Abf, DD, m, kk, lane);
    v16bf b = load_b_frag(Bcol, kk, lane);
    acc = __builtin_amdgcn_wmma_f32_16x16x32_bf16(false, a, false, b,
                                                  (short)0, acc, false, false);
  }
  int h = ncol >> 6, dcol = ncol & 63;
  float bias = b_v[ncol];
  int mhalf = (lane & 16) ? 8 : 0;       // C/D layout: lanes>=16 hold M=8..15
#pragma unroll
  for (int r = 0; r < 8; ++r) {
    int mm = m0 + mhalf + r;
    int bb = mm >> 10, s = mm & (SS - 1);
    vbhT[((size_t)((bb * HH + h) * DK + dcol)) * SS + s] = (__bf16)(acc[r] + bias);
  }
}

// ---- fused: bias copy-out + mask + softmax + PV (x = p @ v per head) ----
// One block per (batch, 16-row query tile). 16 waves: wave r owns row r.
// Loops h = 0..7 reusing the LDS mask tile. p lives only in LDS (bf16).
// Bias stream is b128-vectorized and non-temporal (single-use, 268 MB of the
// ~420 MB total traffic) so it does not evict the re-used vbhT/mask from L2.
__global__ void __launch_bounds__(512)
attn_softmax_pv(const float* __restrict__ bias, const int* __restrict__ mask,
                const __bf16* __restrict__ vbhT, float* __restrict__ out_bias,
                __bf16* __restrict__ x_bf) {
  extern __shared__ char smem[];
  unsigned char* maskb = (unsigned char*)smem;                      // 16*1024   = 16 KB
  float*  wrow = (float*)(smem + 16 * SS);                          // 16*1024*4 = 64 KB
  __bf16* prow = (__bf16*)(smem + 16 * SS + 16 * SS * 4);           // 16*1024*2 = 32 KB

  int lane = threadIdx.x & 31;
  int wave = threadIdx.x >> 5;   // 0..15 == row within tile
  int b = blockIdx.y;
  int q0 = blockIdx.x * 16;
  int r = wave;
  int q = q0 + r;

  const int* mrowp = mask + (size_t)(b * SS + q) * SS;
  for (int i = lane; i < SS; i += 32)
    maskb[r * SS + i] = (unsigned char)(mrowp[i] != 0);
  __syncthreads();

  for (int h = 0; h < HH; ++h) {
    const v4f* brow4 = (const v4f*)(bias + (size_t)((b * HH + h) * SS + q) * SS);
    v4f* obrow4 = (v4f*)(out_bias + (size_t)((b * HH + h) * SS + q) * SS);

    // pass 1: stream bias once (b128, non-temporal) -> copy out + mask + row max
    float tmax = NEGBIG;
    for (int i = lane; i < SS / 4; i += 32) {
      v4f bv = __builtin_nontemporal_load(brow4 + i);
      __builtin_nontemporal_store(bv, obrow4 + i);     // pass-through output
#pragma unroll
      for (int j = 0; j < 4; ++j) {
        int c = i * 4 + j;
        float w = maskb[r * SS + c] ? bv[j] : NEGBIG;
        wrow[r * SS + c] = w;
        tmax = fmaxf(tmax, w);
      }
    }
#pragma unroll
    for (int o = 16; o > 0; o >>= 1) tmax = fmaxf(tmax, __shfl_xor(tmax, o, 32));

    // pass 2: exp + sum (LDS only)
    float tsum = 0.f;
    for (int i = lane; i < SS; i += 32) {
      float e = __expf(wrow[r * SS + i] - tmax);
      wrow[r * SS + i] = e;
      tsum += e;
    }
#pragma unroll
    for (int o = 16; o > 0; o >>= 1) tsum += __shfl_xor(tsum, o, 32);
    float inv = 1.0f / tsum;

    for (int i = lane; i < SS; i += 32)
      prow[r * SS + i] = (__bf16)(wrow[r * SS + i] * inv);
    __syncthreads();

    // PV: x[16x64] = p[16x1024] @ v_bh[1024x64]; 4 waves x one 16x16 tile
    if (wave < 4) {
      int nt = wave * 16;
      int ncol = nt + (lane & 15);
      const __bf16* Bcol = vbhT + (size_t)((b * HH + h) * DK + ncol) * SS;
      v8f acc = {};
#pragma unroll 2
      for (int kk = 0; kk < SS; kk += 32) {
        v16bf a = load_a_frag(prow, SS, (lane & 15), kk, lane);
        v16bf bf = load_b_frag(Bcol, kk, lane);
        acc = __builtin_amdgcn_wmma_f32_16x16x32_bf16(false, a, false, bf,
                                                      (short)0, acc, false, false);
      }
      int mhalf = (lane & 16) ? 8 : 0;
#pragma unroll
      for (int rr = 0; rr < 8; ++rr) {
        int qq = q0 + mhalf + rr;
        x_bf[(size_t)(b * SS + qq) * DD + h * DK + nt + (lane & 15)] = (__bf16)acc[rr];
      }
    }
    __syncthreads();  // protect prow before next head
  }
}

// ---------------- out = x @ W_o^T + b_o (f32 result) ----------------
__global__ void __launch_bounds__(128)
out_gemm(const __bf16* __restrict__ Xbf, const __bf16* __restrict__ Wobf,
         const float* __restrict__ b_o, float* __restrict__ out) {
  int lane = threadIdx.x & 31;
  int wave = threadIdx.x >> 5;
  int m0 = blockIdx.x * 16;
  int ncol = blockIdx.y * 64 + wave * 16 + (lane & 15);
  int m = m0 + (lane & 15);
  const __bf16* Bcol = Wobf + (size_t)ncol * DD;
  v8f acc = {};
#pragma unroll 4
  for (int kk = 0; kk < DD; kk += 32) {
    v16bf a = load_a_frag(Xbf, DD, m, kk, lane);
    v16bf b = load_b_frag(Bcol, kk, lane);
    acc = __builtin_amdgcn_wmma_f32_16x16x32_bf16(false, a, false, b,
                                                  (short)0, acc, false, false);
  }
  float bias = b_o[ncol];
  int mhalf = (lane & 16) ? 8 : 0;
#pragma unroll
  for (int r = 0; r < 8; ++r)   // write-once result: non-temporal store
    __builtin_nontemporal_store(acc[r] + bias,
                                out + (size_t)(m0 + mhalf + r) * DD + ncol);
}

extern "C" void kernel_launch(void* const* d_in, const int* in_sizes, int n_in,
                              void* d_out, int out_size, void* d_ws, size_t ws_size,
                              hipStream_t stream) {
  const float* value = (const float*)d_in[2];
  const float* bias  = (const float*)d_in[3];
  const int*   mask  = (const int*)d_in[4];
  const float* W_v   = (const float*)d_in[5];
  const float* b_v   = (const float*)d_in[6];
  const float* W_o   = (const float*)d_in[7];
  const float* b_o   = (const float*)d_in[8];

  float* out = (float*)d_out;
  float* out_bias = out + (size_t)BB * SS * DD;   // second tuple element

  // workspace layout (13 MB total)
  char* ws = (char*)d_ws;
  __bf16* value_bf = (__bf16*)(ws);                               // 4 MB
  __bf16* Wv_bf    = (__bf16*)(ws + (4u << 20));                  // 512 KB
  __bf16* Wo_bf    = (__bf16*)(ws + (4u << 20) + (512u << 10));   // 512 KB
  __bf16* vbhT     = (__bf16*)(ws + (5u << 20));                  // 4 MB  [B,H,DK,S]
  __bf16* x_bf     = (__bf16*)(ws + (9u << 20));                  // 4 MB  [B*S, D]

  f32_to_bf16<<<2048, 256, 0, stream>>>(value, value_bf, BB * SS * DD);
  f32_to_bf16<<<512,  256, 0, stream>>>(W_v, Wv_bf, DD * DD);
  f32_to_bf16<<<512,  256, 0, stream>>>(W_o, Wo_bf, DD * DD);

  vproj_gemm<<<dim3(BB * SS / 16, DD / 64), 128, 0, stream>>>(value_bf, Wv_bf, b_v, vbhT);

  size_t lds = 16 * SS * (1 + 4 + 2);   // mask(u8)+w(f32)+p(bf16) = 114688 B (<320 KB WGP LDS)
  attn_softmax_pv<<<dim3(SS / 16, BB), 512, lds, stream>>>(bias, mask, vbhT, out_bias, x_bf);

  out_gemm<<<dim3(BB * SS / 16, DD / 64), 128, 0, stream>>>(x_bf, Wo_bf, b_o, out);
}
